// SDGraphEncoder_25383256720092
// MI455X (gfx1250) — matrix operand order repaired and checked by reading, not compiled, not run on hardware
//
#include <hip/hip_runtime.h>
#include <hip/hip_bf16.h>

typedef __attribute__((ext_vector_type(16))) _Float16 v16h;
typedef __attribute__((ext_vector_type(8)))  _Float16 v8h;
typedef __attribute__((ext_vector_type(8)))  float    v8f;

#define BN_S 0.99999500003749965f  /* 1/sqrt(1+1e-5) */

// ---------- WMMA fragment loaders (CDNA5 16x16x32 f16 layouts) ----------
// A (16x32, MxK): lane holds row M=lane&15; contiguous 16 halves at K=(lane>>4)*16
static __device__ __forceinline__ v16h load_a_frag(const _Float16* rowp, int kb, int sel) {
    return *(const v16h*)(rowp + kb + sel * 16);
}
// B (32x16, KxN): lane holds col N=lane&15; contiguous 16 halves at K=(lane>>4)*16
static __device__ __forceinline__ v16h load_b_frag(const _Float16* rowp, int kb, int sel) {
    return *(const v16h*)(rowp + kb + sel * 16);
}

// ---------- weight prep: f16 repacks ----------
__global__ __launch_bounds__(256) void prep_weights_kernel(
        const float* __restrict__ Wdn, const float* __restrict__ Wconv,
        _Float16* __restrict__ W1h, _Float16* __restrict__ Wch, _Float16* __restrict__ Wr) {
    int o = blockIdx.x, c = threadIdx.x;
    float w1 = Wdn[(size_t)o * 512 + c];
    float w2 = Wdn[(size_t)o * 512 + 256 + c];
    W1h[(size_t)o * 256 + c] = (_Float16)w1;
    Wch[(size_t)o * 256 + c] = (_Float16)(w2 - w1);
    for (int dx = 0; dx < 3; ++dx)
        Wr[((size_t)dx * 256 + o) * 256 + c] =
            (_Float16)Wconv[((size_t)o * 256 + c) * 3 + dx];
}

// ---------- build u_sp = concat(sparse, max_p dense)  -> [b][n=32][c=256] f32 ----------
__global__ __launch_bounds__(256) void build_usp_kernel(
        const float* __restrict__ sparse, const float* __restrict__ dense,
        float* __restrict__ usp) {
    int n = blockIdx.x, b = blockIdx.y, c = threadIdx.x;
    float v;
    if (c < 128) {
        v = sparse[((size_t)b * 128 + c) * 32 + n];
    } else {
        const float* dp = dense + (((size_t)b * 128 + (c - 128)) * 32 + n) * 128;
        float m = dp[0];
        for (int p = 1; p < 128; ++p) m = fmaxf(m, dp[p]);
        v = m;
    }
    usp[((size_t)b * 32 + n) * 256 + c] = v;
}

// ---------- build flat dense features: f16 [b][n=4096][c=256] + xx (sum of squares) ----------
__global__ __launch_bounds__(256) void build_flat_kernel(
        const float* __restrict__ sparse, const float* __restrict__ dense,
        _Float16* __restrict__ Xh, float* __restrict__ xx) {
    int n = blockIdx.x, b = blockIdx.y, c = threadIdx.x;
    int s = n >> 7, p = n & 127;
    float v;
    if (c < 128) v = dense[(((size_t)b * 128 + c) * 32 + s) * 128 + p];
    else         v = sparse[((size_t)b * 128 + (c - 128)) * 32 + s];
    Xh[((size_t)b * 4096 + n) * 256 + c] = (_Float16)v;
    __shared__ float red[256];
    red[c] = v * v;
    __syncthreads();
    for (int st = 128; st > 0; st >>= 1) {
        if (c < st) red[c] += red[c + st];
        __syncthreads();
    }
    if (c == 0) xx[(size_t)b * 4096 + n] = red[0];
}

// ---------- sparse kNN (N=32, k=2), scalar via LDS ----------
__global__ __launch_bounds__(256) void knn_sparse_kernel(
        const float* __restrict__ usp, int* __restrict__ idx) {
    int b = blockIdx.x, t = threadIdx.x;
    __shared__ float Xs[32][256];
    __shared__ float xxs[32];
    for (int e = t; e < 32 * 256; e += 256)
        Xs[e >> 8][e & 255] = usp[(size_t)b * 32 * 256 + e];
    __syncthreads();
    if (t < 32) {
        float s = 0.f;
        for (int c = 0; c < 256; ++c) s += Xs[t][c] * Xs[t][c];
        xxs[t] = s;
    }
    __syncthreads();
    if (t < 32) {
        float b0 = -3e38f, b1 = -3e38f; int i0 = 0, i1 = 0;
        for (int j = 0; j < 32; ++j) {
            float d = 0.f;
            for (int c = 0; c < 256; ++c) d += Xs[t][c] * Xs[j][c];
            float nd = 2.f * d - xxs[t] - xxs[j];
            if (nd > b0)      { b1 = b0; i1 = i0; b0 = nd; i0 = j; }
            else if (nd > b1) { b1 = nd; i1 = j; }
        }
        idx[((size_t)b * 32 + t) * 2]     = i0;
        idx[((size_t)b * 32 + t) * 2 + 1] = i1;
    }
}

// ---------- dense kNN: WMMA Gram tiles, fused per-row top-10 (k=10) ----------
__global__ __launch_bounds__(32) void knn_dense_kernel(
        const _Float16* __restrict__ X, const float* __restrict__ xx,
        int* __restrict__ idx) {
    const int N = 4096, C = 256;
    int b = blockIdx.y;
    int mbase = blockIdx.x << 4;
    int lane = threadIdx.x;
    int sel = lane >> 4, lcol = lane & 15;
    const _Float16* Xb = X + (size_t)b * N * C;
    const float* xxb = xx + (size_t)b * N;
    const _Float16* arowp = Xb + (size_t)(mbase + lcol) * C;

    __shared__ float tile[16][17];
    float topv[10]; int topi[10];
#pragma unroll
    for (int t = 0; t < 10; ++t) { topv[t] = -3.0e38f; topi[t] = 0; }

    for (int nbase = 0; nbase < N; nbase += 16) {
        const _Float16* browp = Xb + (size_t)(nbase + lcol) * C;
        v8f acc = {};
#pragma unroll
        for (int kb = 0; kb < C; kb += 32) {
            v16h a  = load_a_frag(arowp, kb, sel);
            v16h bt = load_b_frag(browp, kb, sel);
            acc = __builtin_amdgcn_wmma_f32_16x16x32_f16(false, a, false, bt,
                                                         (short)0, acc, false, false);
        }
        int mhalf = sel * 8;
#pragma unroll
        for (int r = 0; r < 8; ++r) {
            int M = mhalf + r;
            tile[M][lcol] = 2.0f * acc[r] - xxb[mbase + M] - xxb[nbase + lcol];
        }
        __syncthreads();
        if (lane < 16) {
#pragma unroll
            for (int c = 0; c < 16; ++c) {
                float v = tile[lane][c];
                if (v > topv[9]) {
                    int gi = nbase + c;
                    int p = 9;
                    while (p > 0 && topv[p - 1] < v) {
                        topv[p] = topv[p - 1]; topi[p] = topi[p - 1]; --p;
                    }
                    topv[p] = v; topi[p] = gi;
                }
            }
        }
        __syncthreads();
    }
    if (lane < 16) {
        int row = mbase + lane;
#pragma unroll
        for (int t = 0; t < 10; ++t)
            idx[((size_t)b * N + row) * 16 + t] = topi[t];
    }
}

// ---------- generic f16 GEMM: Y[M,N] = A[M,K] * W[N,K]^T (f32 accum) ----------
__global__ __launch_bounds__(32) void gemm_f16_kernel(
        const _Float16* __restrict__ A, const _Float16* __restrict__ W,
        float* __restrict__ Y, int M, int Nn, int Kk) {
    int mbase = blockIdx.x << 4;
    int nbase = blockIdx.y << 4;
    int lane = threadIdx.x;
    int sel = lane >> 4, lcol = lane & 15;
    const _Float16* arowp = A + (size_t)(mbase + lcol) * Kk;
    const _Float16* browp = W + (size_t)(nbase + lcol) * Kk;
    v8f acc = {};
    for (int kb = 0; kb < Kk; kb += 32) {
        __builtin_prefetch(arowp + kb + 128, 0, 0);
        v16h a  = load_a_frag(arowp, kb, sel);
        v16h bt = load_b_frag(browp, kb, sel);
        acc = __builtin_amdgcn_wmma_f32_16x16x32_f16(false, a, false, bt,
                                                     (short)0, acc, false, false);
    }
    int mhalf = sel * 8;
#pragma unroll
    for (int r = 0; r < 8; ++r)
        Y[(size_t)(mbase + mhalf + r) * Nn + nbase + lcol] = acc[r];
}

// ---------- sparse branch linear (tiny, scalar): Y1 = X*W1^T, Yc = X*(W2-W1)^T ----------
__global__ __launch_bounds__(256) void sparse_feat_kernel(
        const float* __restrict__ usp, const float* __restrict__ W,
        float* __restrict__ Y1, float* __restrict__ Yc) {
    int n = blockIdx.x, b = blockIdx.y, o = threadIdx.x;
    __shared__ float xs[256];
    xs[o] = usp[((size_t)b * 32 + n) * 256 + o];
    __syncthreads();
    const float* w = W + (size_t)o * 512;
    float a1 = 0.f, ac = 0.f;
    for (int c = 0; c < 256; ++c) {
        float x = xs[c], w1 = w[c];
        a1 += x * w1;
        ac += x * (w[256 + c] - w1);
    }
    Y1[((size_t)b * 32 + n) * 256 + o] = a1;
    Yc[((size_t)b * 32 + n) * 256 + o] = ac;
}

// ---------- sparse combine: max_k relu(BN(Y1[idx_k] + Yc + b)) -> d_out[0:65536] ----------
__global__ __launch_bounds__(256) void combine_sparse_kernel(
        const float* __restrict__ Y1, const float* __restrict__ Yc,
        const int* __restrict__ idx, const float* __restrict__ bb,
        const float* __restrict__ g, const float* __restrict__ be,
        float* __restrict__ out) {
    int n = blockIdx.x, b = blockIdx.y, o = threadIdx.x;
    float base = Yc[((size_t)b * 32 + n) * 256 + o] + bb[o];
    float gg = g[o] * BN_S, b0 = be[o];
    float m = -3e38f;
    for (int k = 0; k < 2; ++k) {
        int j = idx[((size_t)b * 32 + n) * 2 + k];
        float h = gg * (Y1[((size_t)b * 32 + j) * 256 + o] + base) + b0;
        h = h > 0.f ? h : 0.f;
        m = fmaxf(m, h);
    }
    out[((size_t)b * 256 + o) * 32 + n] = m;
}

// ---------- dense combine: max over k=10 neighbors; write f16 conv input ----------
__global__ __launch_bounds__(256) void combine_dense_kernel(
        const float* __restrict__ Y1, const float* __restrict__ Yc,
        const int* __restrict__ idx, const float* __restrict__ bb,
        const float* __restrict__ g, const float* __restrict__ be,
        _Float16* __restrict__ outh) {
    int n = blockIdx.x, b = blockIdx.y, o = threadIdx.x;
    __shared__ int id[10];
    if (o < 10) id[o] = idx[((size_t)b * 4096 + n) * 16 + o];
    __syncthreads();
    float base = Yc[((size_t)b * 4096 + n) * 256 + o] + bb[o];
    float gg = g[o] * BN_S, b0 = be[o];
    float m = -3e38f;
#pragma unroll
    for (int k = 0; k < 10; ++k) {
        float h = gg * (Y1[((size_t)b * 4096 + id[k]) * 256 + o] + base) + b0;
        h = h > 0.f ? h : 0.f;
        m = fmaxf(m, h);
    }
    outh[((size_t)b * 4096 + n) * 256 + o] = (_Float16)m;
}

// ---------- DownSample conv (1x3, stride 2, pad 1) as 3 accumulated WMMA GEMMs ----------
// Rows m = (b, s, po) over 8*32*64 = 16384; U is point-major [b][n=s*128+p][i] f16.
__global__ __launch_bounds__(32) void conv_down_kernel(
        const _Float16* __restrict__ U, const _Float16* __restrict__ Wr,
        const float* __restrict__ bconv, const float* __restrict__ gc,
        const float* __restrict__ bec, float* __restrict__ out) {
    const int C = 256;
    int mbase = blockIdx.x << 4;
    int nbase = blockIdx.y << 4;
    int lane = threadIdx.x;
    int sel = lane >> 4, lcol = lane & 15;
    int mA = mbase + lcol;                 // A-fragment row for this lane
    int bA = mA >> 11, sA = (mA >> 6) & 31, poA = mA & 63;
    v16h zeroA = {};
    v8f acc = {};
    for (int dx = 0; dx < 3; ++dx) {
        int p = 2 * poA + dx - 1;
        bool ok = (p >= 0) && (p < 128);
        const _Float16* arowp = U + ((size_t)(bA * 4096 + sA * 128 + (ok ? p : 0))) * C;
        const _Float16* browp = Wr + (size_t)dx * C * C + (size_t)(nbase + lcol) * C;
#pragma unroll
        for (int kb = 0; kb < C; kb += 32) {
            v16h a  = ok ? load_a_frag(arowp, kb, sel) : zeroA;
            v16h bt = load_b_frag(browp, kb, sel);
            acc = __builtin_amdgcn_wmma_f32_16x16x32_f16(false, a, false, bt,
                                                         (short)0, acc, false, false);
        }
    }
    int o = nbase + lcol;
    float gg = gc[o] * BN_S, b0 = bec[o], bc = bconv[o];
#pragma unroll
    for (int r = 0; r < 8; ++r) {
        int mr = mbase + sel * 8 + r;
        int b2 = mr >> 11, s2 = (mr >> 6) & 31, po2 = mr & 63;
        float y = gg * (acc[r] + bc) + b0;
        y = y > 0.f ? y : 0.f;
        out[65536 + ((((size_t)b2 * 256 + o) * 32 + s2) << 6) + po2] = y;
    }
}

extern "C" void kernel_launch(void* const* d_in, const int* in_sizes, int n_in,
                              void* d_out, int out_size, void* d_ws, size_t ws_size,
                              hipStream_t stream) {
    const float* sparse = (const float*)d_in[0];
    const float* dense  = (const float*)d_in[1];
    const float* W_sp   = (const float*)d_in[2];
    const float* b_sp   = (const float*)d_in[3];
    const float* g_sp   = (const float*)d_in[4];
    const float* be_sp  = (const float*)d_in[5];
    const float* W_dn   = (const float*)d_in[6];
    const float* b_dn   = (const float*)d_in[7];
    const float* g_dn   = (const float*)d_in[8];
    const float* be_dn  = (const float*)d_in[9];
    const float* W_conv = (const float*)d_in[10];
    const float* b_conv = (const float*)d_in[11];
    const float* g_c    = (const float*)d_in[12];
    const float* be_c   = (const float*)d_in[13];
    float* out = (float*)d_out;

    // bump allocator over workspace
    char* ws = (char*)d_ws;
    size_t off = 0;
    auto alloc = [&](size_t bytes) -> void* {
        void* p = ws + off;
        off += bytes;
        off = (off + 255) & ~(size_t)255;
        return p;
    };
    _Float16* flatT = (_Float16*)alloc((size_t)8 * 4096 * 256 * 2);  // 16 MB
    float*    xx    = (float*)   alloc((size_t)8 * 4096 * 4);
    float*    usp   = (float*)   alloc((size_t)8 * 32 * 256 * 4);
    int*      idxd  = (int*)     alloc((size_t)8 * 4096 * 16 * 4);
    int*      idxs  = (int*)     alloc((size_t)8 * 32 * 2 * 4);
    _Float16* W1h   = (_Float16*)alloc((size_t)256 * 256 * 2);
    _Float16* Wch   = (_Float16*)alloc((size_t)256 * 256 * 2);
    _Float16* Wr    = (_Float16*)alloc((size_t)3 * 256 * 256 * 2);
    float*    Y1d   = (float*)   alloc((size_t)8 * 4096 * 256 * 4); // 32 MB
    float*    Ycd   = (float*)   alloc((size_t)8 * 4096 * 256 * 4); // 32 MB
    _Float16* outh  = (_Float16*)alloc((size_t)8 * 4096 * 256 * 2); // 16 MB
    float*    Ys1   = (float*)   alloc((size_t)8 * 32 * 256 * 4);
    float*    Ysc   = (float*)   alloc((size_t)8 * 32 * 256 * 4);

    prep_weights_kernel<<<256, 256, 0, stream>>>(W_dn, W_conv, W1h, Wch, Wr);
    build_usp_kernel<<<dim3(32, 8), 256, 0, stream>>>(sparse, dense, usp);
    build_flat_kernel<<<dim3(4096, 8), 256, 0, stream>>>(sparse, dense, flatT, xx);

    knn_sparse_kernel<<<8, 256, 0, stream>>>(usp, idxs);
    knn_dense_kernel<<<dim3(256, 8), 32, 0, stream>>>(flatT, xx, idxd);

    sparse_feat_kernel<<<dim3(32, 8), 256, 0, stream>>>(usp, W_sp, Ys1, Ysc);
    combine_sparse_kernel<<<dim3(32, 8), 256, 0, stream>>>(Ys1, Ysc, idxs,
                                                           b_sp, g_sp, be_sp, out);

    gemm_f16_kernel<<<dim3(2048, 16), 32, 0, stream>>>(flatT, W1h, Y1d, 32768, 256, 256);
    gemm_f16_kernel<<<dim3(2048, 16), 32, 0, stream>>>(flatT, Wch, Ycd, 32768, 256, 256);
    combine_dense_kernel<<<dim3(4096, 8), 256, 0, stream>>>(Y1d, Ycd, idxd,
                                                            b_dn, g_dn, be_dn, outh);

    conv_down_kernel<<<dim3(1024, 16), 32, 0, stream>>>(outh, Wr, b_conv, g_c, be_c, out);
}
